// BertSelfAttention_35777077576598
// MI455X (gfx1250) — matrix operand order repaired and checked by reading, not compiled, not run on hardware
//
#include <hip/hip_runtime.h>
#include <stdint.h>

// ---------------------------------------------------------------------------
// BERT self-attention forward on MI455X (gfx1250, wave32, WMMA bf16)
//   B=4, S=2048, H=1024, NH=16, HD=64
// Kernel 1: fused QKV projection (fp32 in -> bf16 Q/K/Vt in workspace)
// Kernel 2: flash attention; K/V tiles staged in LDS via async-to-LDS copies,
//           bf16 WMMA for QK^T and PV, fp32 online softmax.
// ---------------------------------------------------------------------------

#define BB 4
#define SS 2048
#define HH 1024
#define NHH 16
#define HDD 64

typedef __bf16 bf16_t;
typedef __attribute__((ext_vector_type(16))) __bf16 v16bf;
typedef __attribute__((ext_vector_type(8)))  __bf16 v8bf;
typedef __attribute__((ext_vector_type(8)))  float  v8f;
typedef int v4i __attribute__((vector_size(16)));  // b128 payload type

static __device__ __forceinline__ v16bf concat8(v8bf lo, v8bf hi) {
  return __builtin_shufflevector(lo, hi, 0, 1, 2, 3, 4, 5, 6, 7,
                                         8, 9, 10, 11, 12, 13, 14, 15);
}

static __device__ __forceinline__ v16bf ld16(const bf16_t* p0, const bf16_t* p1) {
  return concat8(*reinterpret_cast<const v8bf*>(p0),
                 *reinterpret_cast<const v8bf*>(p1));
}

static __device__ __forceinline__ v8f wmma_bf16(v16bf a, v16bf b, v8f c) {
  // (neg_a, A, neg_b, B, c_mod, C, reuse_a, reuse_b)
  return __builtin_amdgcn_wmma_f32_16x16x32_bf16(false, a, false, b,
                                                 (short)0, c, false, false);
}

// 16-byte global -> LDS asynchronous copy (ASYNCcnt path). Falls back to a
// synchronous copy if the builtin is unavailable on this toolchain.
static __device__ __forceinline__ void async_cp16(const bf16_t* g, bf16_t* l) {
#if __has_builtin(__builtin_amdgcn_global_load_async_to_lds_b128)
  __builtin_amdgcn_global_load_async_to_lds_b128((v4i*)g, (v4i*)l, 0, 0);
#else
  *reinterpret_cast<v8bf*>(l) = *reinterpret_cast<const v8bf*>(g);
#endif
}

static __device__ __forceinline__ void wait_async() {
#if __has_builtin(__builtin_amdgcn_global_load_async_to_lds_b128)
  asm volatile("s_wait_asynccnt 0" ::: "memory");
#endif
}

// ---------------------------------------------------------------------------
// Kernel 1: QKV projection.  grid = (M/128, N/64, 3), block = 256 (8 waves).
// Each wave computes a 16x64 tile of (hidden @ W + b).  K stepped by 32.
// Weights staged transposed in LDS as bf16 (stride 40 to avoid bank conflicts).
// z=0 -> Q [B,NH,S,HD], z=1 -> K [B,NH,S,HD], z=2 -> V transposed [B,NH,HD,S].
// ---------------------------------------------------------------------------
__global__ __launch_bounds__(256) void qkv_proj_kernel(
    const float* __restrict__ hidden,
    const float* __restrict__ Wq, const float* __restrict__ bq,
    const float* __restrict__ Wk, const float* __restrict__ bk,
    const float* __restrict__ Wv, const float* __restrict__ bv,
    bf16_t* __restrict__ Qws, bf16_t* __restrict__ Kws,
    bf16_t* __restrict__ Vtws) {
  const int z = blockIdx.z;
  const float* W    = (z == 0) ? Wq : (z == 1) ? Wk : Wv;
  const float* bias = (z == 0) ? bq : (z == 1) ? bk : bv;

  __shared__ bf16_t Wlds[64][40];  // [n][k], padded

  const int tid  = threadIdx.x;
  const int lane = tid & 31;
  const int w    = tid >> 5;
  const int g    = lane >> 4;   // lane half (0/1)
  const int ln   = lane & 15;
  const int m0   = blockIdx.x * 128 + w * 16;  // 16 rows per wave
  const int n0   = blockIdx.y * 64;            // one full head per y-tile

  v8f acc[4];
#pragma unroll
  for (int t = 0; t < 4; ++t)
#pragma unroll
    for (int r = 0; r < 8; ++r) acc[t][r] = 0.0f;

  for (int k0 = 0; k0 < HH; k0 += 32) {
    __syncthreads();
    // stage W[k0:k0+32][n0:n0+64] transposed as bf16 (native cvt)
    for (int i = tid; i < 32 * 64; i += 256) {
      int kk = i >> 6, nn = i & 63;
      Wlds[nn][kk] = (bf16_t)W[(size_t)(k0 + kk) * HH + n0 + nn];
    }
    __syncthreads();

    // A fragment: row m0+ln, K = g*16 .. g*16+15 contiguous (fp32 -> bf16 packed)
    const float* rowp = hidden + (size_t)(m0 + ln) * HH + k0;
    v8f f0 = *reinterpret_cast<const v8f*>(rowp + g * 16);
    v8f f1 = *reinterpret_cast<const v8f*>(rowp + g * 16 + 8);
    v16bf a = concat8(__builtin_convertvector(f0, v8bf),
                      __builtin_convertvector(f1, v8bf));

#pragma unroll
    for (int t = 0; t < 4; ++t) {
      // B fragment: column n = t*16+ln, K = g*16 .. g*16+15 contiguous in Wlds
      const bf16_t* bp = &Wlds[t * 16 + ln][g * 16];
      v16bf bfrag = ld16(bp, bp + 8);
      acc[t] = wmma_bf16(a, bfrag, acc[t]);
    }
  }

  // epilogue: bias add, convert to bf16, scatter to Q/K/Vt layouts
  const int h = blockIdx.y;  // 64-wide N tile == one head
#pragma unroll
  for (int t = 0; t < 4; ++t) {
    const int d    = t * 16 + ln;
    const float bv_ = bias[n0 + t * 16 + ln];
#pragma unroll
    for (int r = 0; r < 8; ++r) {
      int m  = m0 + r + 8 * g;
      int bb = m >> 11;          // /S
      int ss = m & (SS - 1);
      bf16_t ob = (bf16_t)(acc[t][r] + bv_);
      if (z == 2) {
        Vtws[(((size_t)bb * NHH + h) * HDD + d) * SS + ss] = ob;
      } else {
        bf16_t* dst = (z == 0) ? Qws : Kws;
        dst[(((size_t)bb * NHH + h) * SS + ss) * HDD + d] = ob;
      }
    }
  }
}

// ---------------------------------------------------------------------------
// Kernel 2: flash attention.  grid = (S/64, B*NH), block = 128 (4 waves).
// Each wave owns 16 query rows; loops over 64-key blocks.  K/V tiles for the
// current key block are staged once per block via async global->LDS copies
// (shared by all 4 waves); fragments then come from LDS (conflict-free pad).
// ---------------------------------------------------------------------------
__global__ __launch_bounds__(128) void flash_attn_kernel(
    const bf16_t* __restrict__ Q, const bf16_t* __restrict__ K,
    const bf16_t* __restrict__ Vt, const float* __restrict__ mask,
    float* __restrict__ out) {
  __shared__ bf16_t Ktile[64][72];    // [key][d],  padded (36*ln mod 64 distinct)
  __shared__ bf16_t Vtile[64][72];    // [d][key],  padded
  __shared__ bf16_t Plds[4][16][72];  // per-wave 16x64 P tile, padded

  const int tid  = threadIdx.x;
  const int lane = tid & 31;
  const int w    = tid >> 5;
  const int g    = lane >> 4;
  const int ln   = lane & 15;

  const int bh = blockIdx.y;
  const int b  = bh / NHH;
  const int h  = bh % NHH;
  const bf16_t* Qbh = Q + (size_t)bh * SS * HDD;
  const bf16_t* Kbh = K + (size_t)bh * SS * HDD;
  const bf16_t* Vbh = Vt + (size_t)bh * HDD * SS;
  const int q0 = blockIdx.x * 64 + w * 16;

  // resident Q fragments (two 16x32 A-frags covering HD=64)
  v16bf qf[2];
#pragma unroll
  for (int f = 0; f < 2; ++f) {
    const bf16_t* qp = Qbh + (size_t)(q0 + ln) * HDD + f * 32;
    qf[f] = ld16(qp + g * 16, qp + g * 16 + 8);
  }

  v8f acc_o[4];
  float mrow[8], lrow[8];
#pragma unroll
  for (int t = 0; t < 4; ++t)
#pragma unroll
    for (int r = 0; r < 8; ++r) acc_o[t][r] = 0.0f;
#pragma unroll
  for (int r = 0; r < 8; ++r) { mrow[r] = -1e30f; lrow[r] = 0.0f; }

  for (int kb = 0; kb < SS; kb += 64) {
    // ---- stage K[kb:kb+64][0:64] and Vt[0:64][kb:kb+64] into LDS (async) ----
    __syncthreads();  // previous iteration's readers are done
#pragma unroll
    for (int i = tid; i < 64 * 8; i += 128) {  // 512 x 16-byte chunks per tile
      int row = i >> 3, c8 = (i & 7) * 8;
      async_cp16(Kbh + (size_t)(kb + row) * HDD + c8, &Ktile[row][c8]);
      async_cp16(Vbh + (size_t)row * SS + kb + c8, &Vtile[row][c8]);
    }
    wait_async();
    __syncthreads();

    // ---- scores: S = Q K^T * (1/8) + mask ----
    v8f sa[4];
#pragma unroll
    for (int t = 0; t < 4; ++t) {
#pragma unroll
      for (int r = 0; r < 8; ++r) sa[t][r] = 0.0f;
#pragma unroll
      for (int f = 0; f < 2; ++f) {
        const bf16_t* kp = &Ktile[t * 16 + ln][f * 32 + g * 16];
        v16bf kfrag = ld16(kp, kp + 8);
        sa[t] = wmma_bf16(qf[f], kfrag, sa[t]);
      }
    }
#pragma unroll
    for (int t = 0; t < 4; ++t) {
      float mk = mask[(size_t)b * SS + kb + t * 16 + ln];
#pragma unroll
      for (int r = 0; r < 8; ++r) sa[t][r] = sa[t][r] * 0.125f + mk;
    }

    // ---- online softmax (row stats per lane-half; reduce over 16 lanes) ----
    float mnew[8], alpha[8], rs[8];
#pragma unroll
    for (int r = 0; r < 8; ++r) {
      float v = fmaxf(fmaxf(sa[0][r], sa[1][r]), fmaxf(sa[2][r], sa[3][r]));
      v = fmaxf(v, __shfl_xor(v, 1, 32));
      v = fmaxf(v, __shfl_xor(v, 2, 32));
      v = fmaxf(v, __shfl_xor(v, 4, 32));
      v = fmaxf(v, __shfl_xor(v, 8, 32));
      mnew[r]  = fmaxf(mrow[r], v);
      alpha[r] = __expf(mrow[r] - mnew[r]);
      rs[r]    = 0.0f;
    }
#pragma unroll
    for (int t = 0; t < 4; ++t)
#pragma unroll
      for (int r = 0; r < 8; ++r) {
        float p = __expf(sa[t][r] - mnew[r]);
        sa[t][r] = p;
        rs[r] += p;
      }
#pragma unroll
    for (int r = 0; r < 8; ++r) {
      rs[r] += __shfl_xor(rs[r], 1, 32);
      rs[r] += __shfl_xor(rs[r], 2, 32);
      rs[r] += __shfl_xor(rs[r], 4, 32);
      rs[r] += __shfl_xor(rs[r], 8, 32);
      lrow[r] = lrow[r] * alpha[r] + rs[r];
      mrow[r] = mnew[r];
    }
#pragma unroll
    for (int t = 0; t < 4; ++t)
#pragma unroll
      for (int r = 0; r < 8; ++r) acc_o[t][r] *= alpha[r];

    // ---- P (C-layout) -> LDS -> A-fragment layout (native bf16 cvt) ----
#pragma unroll
    for (int t = 0; t < 4; ++t)
#pragma unroll
      for (int r = 0; r < 8; ++r)
        Plds[w][r + 8 * g][t * 16 + ln] = (bf16_t)sa[t][r];
    asm volatile("s_wait_dscnt 0" ::: "memory");  // per-wave LDS RAW fence

    // ---- PV: acc_o += P @ V  (V tile is [d][key] in LDS) ----
#pragma unroll
    for (int f = 0; f < 2; ++f) {
      const bf16_t* pp = &Plds[w][ln][f * 32];
      v16bf pf = ld16(pp + g * 16, pp + g * 16 + 8);
#pragma unroll
      for (int t = 0; t < 4; ++t) {
        const bf16_t* vp = &Vtile[t * 16 + ln][f * 32 + g * 16];
        v16bf vfrag = ld16(vp, vp + 8);
        acc_o[t] = wmma_bf16(pf, vfrag, acc_o[t]);
      }
    }
  }

  // ---- normalize and store ctx -> [B,S,H] fp32 ----
#pragma unroll
  for (int t = 0; t < 4; ++t) {
    const int d = t * 16 + ln;
#pragma unroll
    for (int r = 0; r < 8; ++r) {
      int sq = q0 + r + 8 * g;
      out[((size_t)b * SS + sq) * HH + h * 64 + d] = acc_o[t][r] / lrow[r];
    }
  }
}

// ---------------------------------------------------------------------------
extern "C" void kernel_launch(void* const* d_in, const int* in_sizes, int n_in,
                              void* d_out, int out_size, void* d_ws,
                              size_t ws_size, hipStream_t stream) {
  const float* hidden = (const float*)d_in[0];
  const float* mask   = (const float*)d_in[1];
  const float* Wq     = (const float*)d_in[2];
  const float* bq     = (const float*)d_in[3];
  const float* Wk     = (const float*)d_in[4];
  const float* bk     = (const float*)d_in[5];
  const float* Wv     = (const float*)d_in[6];
  const float* bv     = (const float*)d_in[7];
  float* out = (float*)d_out;

  const size_t elems = (size_t)BB * NHH * SS * HDD;  // 8,388,608 per tensor
  bf16_t* Qws  = (bf16_t*)d_ws;
  bf16_t* Kws  = Qws + elems;
  bf16_t* Vtws = Kws + elems;

  dim3 g1((BB * SS) / 128, HH / 64, 3);
  qkv_proj_kernel<<<g1, 256, 0, stream>>>(hidden, Wq, bq, Wk, bk, Wv, bv,
                                          Qws, Kws, Vtws);

  dim3 g2(SS / 64, BB * NHH);
  flash_attn_kernel<<<g2, 128, 0, stream>>>(Qws, Kws, Vtws, mask, out);
}
